// _MoEBlock_2499670966569
// MI455X (gfx1250) — matrix operand order repaired and checked by reading, not compiled, not run on hardware
//
#include <hip/hip_runtime.h>

#define Hd   1024
#define Ed   8
#define Td   8192          // B*S
#define CAP  1024          // Td / Ed

typedef __attribute__((ext_vector_type(16))) _Float16 v16h;
typedef __attribute__((ext_vector_type(8)))  _Float16 v8h;
typedef __attribute__((ext_vector_type(4)))  _Float16 v4h;
typedef __attribute__((ext_vector_type(8)))  float    v8f;

// ---------------------------------------------------------------- zero output
__global__ void zero_out_kernel(float4* __restrict__ out) {
    float4 z = make_float4(0.f, 0.f, 0.f, 0.f);
    out[(size_t)blockIdx.x * blockDim.x + threadIdx.x] = z;
}

// ---------------------------------------------------------------- gating
// one wave32 per token: logits = x[t] @ wg, top-1 expert + softmax prob
__global__ void __launch_bounds__(256)
gate_kernel(const float* __restrict__ x, const float* __restrict__ wg,
            int* __restrict__ tok_expert, float* __restrict__ tok_prob) {
    const int lane = threadIdx.x & 31;
    const int t = blockIdx.x * 8 + (threadIdx.x >> 5);
    const float* xr = x + (size_t)t * Hd;

    float acc[Ed];
#pragma unroll
    for (int e = 0; e < Ed; ++e) acc[e] = 0.f;

    for (int h = lane; h < Hd; h += 32) {
        float xv = xr[h];
        const float4* wv = (const float4*)(wg + (size_t)h * Ed);
        float4 w0 = wv[0], w1 = wv[1];
        acc[0] += xv * w0.x; acc[1] += xv * w0.y;
        acc[2] += xv * w0.z; acc[3] += xv * w0.w;
        acc[4] += xv * w1.x; acc[5] += xv * w1.y;
        acc[6] += xv * w1.z; acc[7] += xv * w1.w;
    }
#pragma unroll
    for (int off = 16; off > 0; off >>= 1)
#pragma unroll
        for (int e = 0; e < Ed; ++e) acc[e] += __shfl_xor(acc[e], off, 32);

    if (lane == 0) {
        float m = acc[0]; int am = 0;
#pragma unroll
        for (int e = 1; e < Ed; ++e) if (acc[e] > m) { m = acc[e]; am = e; }
        float s = 0.f;
#pragma unroll
        for (int e = 0; e < Ed; ++e) s += __expf(acc[e] - m);
        tok_expert[t] = am;
        tok_prob[t]   = 1.f / s;   // softmax prob of the argmax expert
    }
}

// ---------------------------------------------------------------- routing
// single workgroup: deterministic token-order scan (matches reference cumsum)
__global__ void __launch_bounds__(1024)
route_kernel(const int* __restrict__ tok_expert, const float* __restrict__ tok_prob,
             int* __restrict__ slot_token, float* __restrict__ slot_gate) {
    __shared__ int sc[1024 * Ed];                 // 32 KB
    const int tid  = threadIdx.x;
    const int base = tid * 8;                     // 8 tokens per thread

    // init slot->token map to "empty"
#pragma unroll
    for (int j = 0; j < 8; ++j) slot_token[base + j] = -1;

    int cnt[Ed];
#pragma unroll
    for (int e = 0; e < Ed; ++e) cnt[e] = 0;
    int ex[8];
#pragma unroll
    for (int j = 0; j < 8; ++j) { ex[j] = tok_expert[base + j]; cnt[ex[j]]++; }
#pragma unroll
    for (int e = 0; e < Ed; ++e) sc[tid * Ed + e] = cnt[e];
    __syncthreads();

    // inclusive Hillis-Steele scan over 1024 thread-chunks, per expert
    for (int off = 1; off < 1024; off <<= 1) {
        int v[Ed];
        if (tid >= off)
#pragma unroll
            for (int e = 0; e < Ed; ++e) v[e] = sc[(tid - off) * Ed + e];
        __syncthreads();
        if (tid >= off)
#pragma unroll
            for (int e = 0; e < Ed; ++e) sc[tid * Ed + e] += v[e];
        __syncthreads();
    }

    int pos[Ed];
#pragma unroll
    for (int e = 0; e < Ed; ++e) pos[e] = sc[tid * Ed + e] - cnt[e];  // exclusive

#pragma unroll
    for (int j = 0; j < 8; ++j) {
        int e = ex[j];
        int p = pos[e]++;
        if (p < CAP) {
            slot_token[e * CAP + p] = base + j;
            slot_gate [e * CAP + p] = tok_prob[base + j];
        }
    }
}

// ---------------------------------------------------------------- dispatch (gather + f32->f16)
__global__ void __launch_bounds__(256)
dispatch_kernel(const float* __restrict__ x, const int* __restrict__ slot_token,
                _Float16* __restrict__ exp_in) {
    const int s = blockIdx.x;            // slot id in [0, E*CAP)
    const int t = slot_token[s];
    const int j = threadIdx.x;           // 4 elements each (H = 256*4)
    v4h o;
    if (t >= 0) {
        float4 v = ((const float4*)(x + (size_t)t * Hd))[j];
        o[0] = (_Float16)v.x; o[1] = (_Float16)v.y;
        o[2] = (_Float16)v.z; o[3] = (_Float16)v.w;
    } else {
        o[0] = o[1] = o[2] = o[3] = (_Float16)0.f;
    }
    ((v4h*)(exp_in + (size_t)s * Hd))[j] = o;
}

// ---------------------------------------------------------------- W f32->f16
__global__ void __launch_bounds__(256)
cvt_w_kernel(const float* __restrict__ W, _Float16* __restrict__ Wh) {
    size_t i = (size_t)blockIdx.x * blockDim.x + threadIdx.x;
    float4 v = ((const float4*)W)[i];
    v4h o;
    o[0] = (_Float16)v.x; o[1] = (_Float16)v.y;
    o[2] = (_Float16)v.z; o[3] = (_Float16)v.w;
    ((v4h*)Wh)[i] = o;
}

// ---------------------------------------------------------------- WMMA fragment loads
// A (16x32 f16, ISA 7.12.2): lane m / lane m+16 hold K {0-7,16-23} / {8-15,24-31}
__device__ __forceinline__ v16h load_fragA(const _Float16* p) {
    v8h lo = *(const v8h*)(p);        // K k..k+7
    v8h hi = *(const v8h*)(p + 16);   // K k+16..k+23
    v16h r;
#pragma unroll
    for (int i = 0; i < 8; ++i) { r[i] = lo[i]; r[i + 8] = hi[i]; }
    return r;
}
// B (32x16 f16): lane n / lane n+16 hold K {0-7,16-23} / {8-15,24-31} of col n
__device__ __forceinline__ v16h load_fragB(const _Float16* p) {
    v8h lo = *(const v8h*)(p);
    v8h hi = *(const v8h*)(p + 16);
    v16h r;
#pragma unroll
    for (int i = 0; i < 8; ++i) { r[i] = lo[i]; r[i + 8] = hi[i]; }
    return r;
}

// ---------------------------------------------------------------- expert GEMM + combine
// C[e] (CAP x H) = exp_in[e] (CAP x H_in) @ W[e]^T, f32 accum; epilogue: +bias,
// scale by gate prob, scatter row -> out[token]. 8 waves/block, 128x128 tile.
__global__ void __launch_bounds__(256)
moe_gemm_kernel(const _Float16* __restrict__ A,   // [E, CAP, H]
                const _Float16* __restrict__ Bw,  // [E, H, H] (row n = W[e][n][:])
                const float* __restrict__ bias,   // [E, H]
                const int* __restrict__ slot_token,
                const float* __restrict__ slot_gate,
                float* __restrict__ out) {        // [T, H]
    const int lane = threadIdx.x & 31;
    const int w    = threadIdx.x >> 5;            // 8 waves
    const int e    = blockIdx.x >> 6;             // expert
    const int tile = blockIdx.x & 63;
    const int m0   = (tile >> 3) * 128 + (w & 3) * 32;   // 32-row strip per wave
    const int n0   = (tile & 7) * 128 + (w >> 2) * 64;   // 64-col strip per wave

    const _Float16* Ae = A  + (size_t)e * CAP * Hd;
    const _Float16* Be = Bw + (size_t)e * Hd * Hd;

    const int lm = lane & 15;
    const int kh = lane >> 4;                      // lane half

    v8f acc[2][4];
#pragma unroll
    for (int mi = 0; mi < 2; ++mi)
#pragma unroll
        for (int ni = 0; ni < 4; ++ni) {
            v8f z = {};
            acc[mi][ni] = z;
        }

    const _Float16* aP0 = Ae + (size_t)(m0 + lm) * Hd + kh * 8;
    const _Float16* aP1 = aP0 + 16 * Hd;
    const _Float16* bP[4];
    bP[0] = Be + (size_t)(n0 + lm) * Hd + kh * 8;
#pragma unroll
    for (int ni = 1; ni < 4; ++ni) bP[ni] = bP[0] + (size_t)ni * 16 * Hd;

    for (int k = 0; k < Hd; k += 32) {
        __builtin_prefetch(aP0 + k + 256, 0, 1);   // stream-ahead hints
        __builtin_prefetch(bP[0] + k + 256, 0, 1);

        v16h a0 = load_fragA(aP0 + k);
        v16h a1 = load_fragA(aP1 + k);
        v16h b0 = load_fragB(bP[0] + k);
        v16h b1 = load_fragB(bP[1] + k);
        v16h b2 = load_fragB(bP[2] + k);
        v16h b3 = load_fragB(bP[3] + k);

        acc[0][0] = __builtin_amdgcn_wmma_f32_16x16x32_f16(false, a0, false, b0, (short)0, acc[0][0], false, false);
        acc[0][1] = __builtin_amdgcn_wmma_f32_16x16x32_f16(false, a0, false, b1, (short)0, acc[0][1], false, false);
        acc[0][2] = __builtin_amdgcn_wmma_f32_16x16x32_f16(false, a0, false, b2, (short)0, acc[0][2], false, false);
        acc[0][3] = __builtin_amdgcn_wmma_f32_16x16x32_f16(false, a0, false, b3, (short)0, acc[0][3], false, false);
        acc[1][0] = __builtin_amdgcn_wmma_f32_16x16x32_f16(false, a1, false, b0, (short)0, acc[1][0], false, false);
        acc[1][1] = __builtin_amdgcn_wmma_f32_16x16x32_f16(false, a1, false, b1, (short)0, acc[1][1], false, false);
        acc[1][2] = __builtin_amdgcn_wmma_f32_16x16x32_f16(false, a1, false, b2, (short)0, acc[1][2], false, false);
        acc[1][3] = __builtin_amdgcn_wmma_f32_16x16x32_f16(false, a1, false, b3, (short)0, acc[1][3], false, false);
    }

    // epilogue: bias, gate scale, scatter to out[token]
    const int*   stok  = slot_token + e * CAP;
    const float* sgate = slot_gate  + e * CAP;
    const float* be    = bias + (size_t)e * Hd;

#pragma unroll
    for (int mi = 0; mi < 2; ++mi) {
        const int rbase = m0 + mi * 16 + kh * 8;   // slot row of VGPR i is rbase+i
        int   tarr[8];
        float garr[8];
#pragma unroll
        for (int i = 0; i < 8; ++i) {
            tarr[i] = stok[rbase + i];
            garr[i] = sgate[rbase + i];
        }
#pragma unroll
        for (int ni = 0; ni < 4; ++ni) {
            const int col = n0 + ni * 16 + lm;
            const float bb = be[col];
#pragma unroll
            for (int i = 0; i < 8; ++i) {
                if (tarr[i] >= 0)
                    out[(size_t)tarr[i] * Hd + col] = garr[i] * (acc[mi][ni][i] + bb);
            }
        }
    }
}

// ---------------------------------------------------------------- launch
extern "C" void kernel_launch(void* const* d_in, const int* in_sizes, int n_in,
                              void* d_out, int out_size, void* d_ws, size_t ws_size,
                              hipStream_t stream) {
    const float* x  = (const float*)d_in[0];   // [B,S,H]
    const float* wg = (const float*)d_in[1];   // [H,E]
    const float* W  = (const float*)d_in[2];   // [E,H,H]
    const float* b  = (const float*)d_in[3];   // [E,H]
    float* out = (float*)d_out;                // [B,S,H]

    char* ws = (char*)d_ws;
    int*      tok_expert = (int*)      (ws);                         // T ints
    float*    tok_prob   = (float*)    (ws + 32768);                 // T floats
    int*      slot_token = (int*)      (ws + 65536);                 // E*CAP ints
    float*    slot_gate  = (float*)    (ws + 98304);                 // E*CAP floats
    _Float16* exp_in_h   = (_Float16*) (ws + 131072);                // E*CAP*H f16 = 16MB
    _Float16* w_h        = (_Float16*) (ws + 131072 + 16777216);     // E*H*H  f16 = 16MB

    // 1. zero output (dropped tokens stay zero)
    zero_out_kernel<<<Td * Hd / (4 * 256), 256, 0, stream>>>((float4*)out);
    // 2. gating: top-1 expert + prob per token
    gate_kernel<<<Td / 8, 256, 0, stream>>>(x, wg, tok_expert, tok_prob);
    // 3. deterministic capacity routing
    route_kernel<<<1, 1024, 0, stream>>>(tok_expert, tok_prob, slot_token, slot_gate);
    // 4. gather tokens into per-expert f16 buffers
    dispatch_kernel<<<Ed * CAP, 256, 0, stream>>>(x, slot_token, exp_in_h);
    // 5. convert expert weights to f16
    cvt_w_kernel<<<Ed * Hd * Hd / (4 * 256), 256, 0, stream>>>(W, w_h);
    // 6. expert GEMMs (WMMA) + bias + gate scale + scatter
    moe_gemm_kernel<<<Ed * 64, 256, 0, stream>>>(exp_in_h, w_h, b,
                                                 slot_token, slot_gate, out);
}
